// CausalWanSelfAttention_6193342841680
// MI455X (gfx1250) — compile-verified, matrix operand order checked by reading, not run on hardware
//
#include <hip/hip_runtime.h>
#include <hip/hip_bf16.h>
#include <stdint.h>

#define SEQ   1040
#define DIM   2048
#define NH    16
#define HD    128
#define LPAD  9376           // CACHE(8320)+SEQ(1040)=9360, padded to mult of 32
#define QT    (SEQ / 16)     // 65 query tiles

#if defined(__has_builtin)
#  if __has_builtin(__builtin_amdgcn_global_load_async_to_lds_b128)
#    define HAVE_ASYNC 1
#  endif
#endif
#ifndef HAVE_ASYNC
#  define HAVE_ASYNC 0
#endif

typedef __attribute__((ext_vector_type(16))) __bf16       bf16x16;
typedef __attribute__((ext_vector_type(8)))  float        f32x8;
typedef __attribute__((ext_vector_type(8)))  unsigned int u32x8;
typedef __attribute__((ext_vector_type(4)))  int          i32x4;

union FragU { u32x8 u; bf16x16 b; };

__device__ __forceinline__ unsigned short f2bf(float f) {
  unsigned u = __builtin_bit_cast(unsigned, f);
  u += 0x7FFFu + ((u >> 16) & 1u);             // round-to-nearest-even
  return (unsigned short)(u >> 16);
}
__device__ __forceinline__ unsigned pack2bf(float lo, float hi) {
  return (unsigned)f2bf(lo) | ((unsigned)f2bf(hi) << 16);
}

// Load a 16x32 bf16 WMMA fragment (A-style; identical mapping serves the B
// operand when the source is stored [N][K] row-major).
// ISA 7.12.2: lane(0..15)=row, lanes 16..31 same rows, K split:
//   vgpr i<4 : K = half*8 + 2i{,+1};  i>=4 : K = 16 + half*8 + 2(i-4){,+1}
__device__ __forceinline__ bf16x16 load_frag(const unsigned short* __restrict__ base,
                                             int stride, int row, int k0, int half) {
  FragU f;
  const unsigned short* p = base + (size_t)row * (size_t)stride + k0;
#pragma unroll
  for (int i = 0; i < 8; ++i) {
    int k = ((i >> 2) << 4) + half * 8 + ((i & 3) << 1);
    f.u[i] = *(const unsigned*)(p + k);
  }
  return f.b;
}

__device__ __forceinline__ f32x8 wmma_bf16(bf16x16 a, bf16x16 b, f32x8 c) {
  return __builtin_amdgcn_wmma_f32_16x16x32_bf16(false, a, false, b, (short)0, c,
                                                 false, false);
}

// Async-copy one 16x64-element bf16 A-chunk (2KB) into LDS; called by wave 0.
// Each of 4 b128 ops moves 512B across 32 lanes. LDS layout: row-major,
// 64 elements (128B) per row.
__device__ __forceinline__ void stage_a(const unsigned short* __restrict__ A,
                                        unsigned short* lds_tile, int tid,
                                        size_t gbase_bytes, int K) {
#pragma unroll
  for (int j = 0; j < 4; ++j) {
    int boff = j * 512 + tid * 16;
    int row = boff >> 7, col = boff & 127;
    const char* src = (const char*)A + gbase_bytes + (size_t)row * K * 2 + col;
    char* dst = (char*)lds_tile + boff;
#if HAVE_ASYNC
    __builtin_amdgcn_global_load_async_to_lds_b128((i32x4*)src, (i32x4*)dst, 0, 0);
#else
    uint4 d = *(const uint4*)src;
    *(uint4*)dst = d;
#endif
  }
}

__device__ __forceinline__ void wait_async() {
#if HAVE_ASYNC
#  if __has_builtin(__builtin_amdgcn_s_wait_asynccnt)
  __builtin_amdgcn_s_wait_asynccnt(0);
#  else
  asm volatile("s_wait_asynccnt 0x0" ::: "memory");
#  endif
#endif
}

// Read an A fragment (16x32 sub-chunk) out of the staged LDS tile.
__device__ __forceinline__ bf16x16 lds_frag(const unsigned short* tile, int lane,
                                            int half, int ks) {
  FragU f;
  const unsigned short* lr = tile + lane * 64 + ks * 32 + half * 8;
  uint4 r0 = *(const uint4*)lr;
  uint4 r1 = *(const uint4*)(lr + 16);
  f.u[0] = r0.x; f.u[1] = r0.y; f.u[2] = r0.z; f.u[3] = r0.w;
  f.u[4] = r1.x; f.u[5] = r1.y; f.u[6] = r1.z; f.u[7] = r1.w;
  return f.b;
}

// ---------------- conversion / layout kernels ----------------

__global__ void k_cvt_x(const float* __restrict__ x, unsigned short* __restrict__ xb, int n) {
  int i = blockIdx.x * blockDim.x + threadIdx.x;
  if (i < n) xb[i] = f2bf(x[i]);
}

// Wt[n][k] = bf16(W[k][n])  (so GEMM B operand streams as contiguous K pairs)
__global__ void k_transpose_w(const float* __restrict__ W, unsigned short* __restrict__ Wt) {
  int i = blockIdx.x * blockDim.x + threadIdx.x;
  if (i >= DIM * DIM) return;
  int nn = i >> 11, k = i & (DIM - 1);
  Wt[i] = f2bf(W[(size_t)k * DIM + nn]);
}

// Convert cached KV into padded bf16 buffers; zero-fill the pad tail.
// K kept [pos][h*128+d]; V transposed to [h*128+d][pos].
__global__ void k_cache(const float* __restrict__ ck, const float* __restrict__ cv,
                        unsigned short* __restrict__ kf, unsigned short* __restrict__ vT,
                        const int* __restrict__ p_start, const int* __restrict__ p_len) {
  int i = blockIdx.x * blockDim.x + threadIdx.x;
  if (i >= LPAD * DIM) return;
  int pos = i >> 11, e = i & (DIM - 1);
  int kvs = p_start[0], kvl = p_len[0];
  int L = kvl + SEQ;
  if (pos < kvl) {
    size_t src = (size_t)(kvs + pos) * DIM + e;
    kf[(size_t)pos * DIM + e]  = f2bf(ck[src]);
    vT[(size_t)e * LPAD + pos] = f2bf(cv[src]);
  } else if (pos >= L) {
    kf[(size_t)pos * DIM + e]  = 0;
    vT[(size_t)e * LPAD + pos] = 0;
  }
}

// ---------------- bf16 WMMA GEMM: C[M,N] = A[M,K] * Bt[N,K]^T + bias ----------------
// Block: 8 waves, one 16-row M tile, 32 N-tiles. A-chunks (16x64) are
// double-buffered in LDS via async global->LDS copies issued by wave 0.
__global__ void k_gemm(const unsigned short* __restrict__ A,
                       const unsigned short* __restrict__ Bt,
                       const float* __restrict__ bias, float* __restrict__ C,
                       int N, int K) {
  __shared__ unsigned short atile[2][16 * 64];   // 2 x 2KB double buffer
  int tid = threadIdx.x & 31, wave = threadIdx.x >> 5;
  int half = tid >> 4, lane = tid & 15;
  int mt  = blockIdx.y;
  int nt0 = (blockIdx.x * 8 + wave) * 4;
  f32x8 zero = {0.f, 0.f, 0.f, 0.f, 0.f, 0.f, 0.f, 0.f};
  f32x8 acc[4];
#pragma unroll
  for (int t = 0; t < 4; ++t) acc[t] = zero;

  const int nch = K / 64;
  size_t abase = (size_t)mt * 16 * K * 2;        // byte offset of tile (row0, col0)

  if (wave == 0) { stage_a(A, atile[0], tid, abase, K); wait_async(); }
  __syncthreads();

  for (int c = 0; c < nch; ++c) {
    int buf = c & 1;
    if (wave == 0 && c + 1 < nch)
      stage_a(A, atile[buf ^ 1], tid, abase + (size_t)(c + 1) * 64 * 2, K);
#pragma unroll
    for (int ks = 0; ks < 2; ++ks) {
      int k0 = c * 64 + ks * 32;
      bf16x16 b[4];
#pragma unroll
      for (int t = 0; t < 4; ++t)
        b[t] = load_frag(Bt, K, (nt0 + t) * 16 + lane, k0, half);
      bf16x16 a = lds_frag(atile[buf], lane, half, ks);
#pragma unroll
      for (int t = 0; t < 4; ++t)
        acc[t] = wmma_bf16(a, b[t], acc[t]);
    }
    if (wave == 0) wait_async();
    __syncthreads();
  }
#pragma unroll
  for (int t = 0; t < 4; ++t) {
    int col = (nt0 + t) * 16 + lane;
    float bia = bias[col];
#pragma unroll
    for (int r = 0; r < 8; ++r) {
      int row = mt * 16 + r + 8 * half;          // C layout: lane=N, vgpr r=M
      C[(size_t)row * N + col] = acc[t][r] + bia;
    }
  }
}

// ---------------- fused RMSNorm (full dim) + RoPE + KV append ----------------
__global__ void k_norm_rope(const float* __restrict__ Q, const float* __restrict__ Kp,
                            const float* __restrict__ V,
                            const float* __restrict__ gq, const float* __restrict__ gk,
                            const float* __restrict__ freqs,
                            unsigned short* __restrict__ qb, unsigned short* __restrict__ kf,
                            unsigned short* __restrict__ vT, const int* __restrict__ p_len) {
  __shared__ float red[256];
  __shared__ float rq_s, rk_s;
  int s = blockIdx.x, t = threadIdx.x;
  const float* qrow = Q  + (size_t)s * DIM;
  const float* krow = Kp + (size_t)s * DIM;
  const float* vrow = V  + (size_t)s * DIM;
  float q8[8], k8[8], v8[8];
  float sq = 0.f, sk = 0.f;
#pragma unroll
  for (int j = 0; j < 8; ++j) {
    q8[j] = qrow[t * 8 + j]; k8[j] = krow[t * 8 + j]; v8[j] = vrow[t * 8 + j];
    sq += q8[j] * q8[j];     sk += k8[j] * k8[j];
  }
  red[t] = sq; __syncthreads();
  for (int o = 128; o > 0; o >>= 1) { if (t < o) red[t] += red[t + o]; __syncthreads(); }
  if (t == 0) rq_s = rsqrtf(red[0] / (float)DIM + 1e-6f);
  __syncthreads();
  red[t] = sk; __syncthreads();
  for (int o = 128; o > 0; o >>= 1) { if (t < o) red[t] += red[t + o]; __syncthreads(); }
  if (t == 0) rk_s = rsqrtf(red[0] / (float)DIM + 1e-6f);
  __syncthreads();
  float rq = rq_s, rk = rk_s;
  int kvl = p_len[0];
#pragma unroll
  for (int pp = 0; pp < 4; ++pp) {
    int e  = t * 4 + pp;                // global rotation-pair index
    int e0 = 2 * e, e1 = 2 * e + 1;
    float ang = freqs[s * (HD / 2) + (e & 63)];
    float c = __cosf(ang), sn = __sinf(ang);
    float qe = q8[2 * pp] * rq * gq[e0], qo = q8[2 * pp + 1] * rq * gq[e1];
    float ke = k8[2 * pp] * rk * gk[e0], ko = k8[2 * pp + 1] * rk * gk[e1];
    qb[(size_t)s * DIM + e0] = f2bf(qe * c - qo * sn);
    qb[(size_t)s * DIM + e1] = f2bf(qe * sn + qo * c);
    size_t kro = (size_t)(kvl + s) * DIM;
    kf[kro + e0] = f2bf(ke * c - ko * sn);
    kf[kro + e1] = f2bf(ke * sn + ko * c);
    vT[(size_t)e0 * LPAD + kvl + s] = f2bf(v8[2 * pp]);
    vT[(size_t)e1 * LPAD + kvl + s] = f2bf(v8[2 * pp + 1]);
  }
}

// ---------------- flash attention: one wave per (head, 16-query tile) ----------------
__global__ void k_attn(const unsigned short* __restrict__ qb,
                       const unsigned short* __restrict__ kf,
                       const unsigned short* __restrict__ vT,
                       unsigned short* __restrict__ ob,
                       const int* __restrict__ p_len) {
  __shared__ float plds[4][16 * 32];
  int tid = threadIdx.x & 31, wave = threadIdx.x >> 5;
  int half = tid >> 4, lane = tid & 15;
  int h  = blockIdx.x;
  int qt = blockIdx.y * 4 + wave;
  if (qt >= QT) return;                         // wave-uniform exit (EXEC stays full)
  int L = p_len[0] + SEQ;
  const float scale = 0.08838834764831845f;     // 1/sqrt(128)
  float* lds = plds[wave];

  bf16x16 qfrag[4];
#pragma unroll
  for (int kk = 0; kk < 4; ++kk)
    qfrag[kk] = load_frag(qb + h * HD, DIM, qt * 16 + lane, kk * 32, half);

  f32x8 zero = {0.f, 0.f, 0.f, 0.f, 0.f, 0.f, 0.f, 0.f};
  f32x8 o[8];
#pragma unroll
  for (int t = 0; t < 8; ++t) o[t] = zero;
  float mrow[8], lrow[8];
#pragma unroll
  for (int r = 0; r < 8; ++r) { mrow[r] = -3.0e38f; lrow[r] = 0.f; }

  for (int base = 0; base < L; base += 32) {
    // Issue all 8 K fragments, then burst the score WMMAs.
    bf16x16 kb[8];
#pragma unroll
    for (int kk = 0; kk < 4; ++kk) {
      kb[kk]     = load_frag(kf + h * HD, DIM, base + lane,      kk * 32, half);
      kb[4 + kk] = load_frag(kf + h * HD, DIM, base + 16 + lane, kk * 32, half);
    }
    f32x8 s0 = zero, s1 = zero;
#pragma unroll
    for (int kk = 0; kk < 4; ++kk) s0 = wmma_bf16(qfrag[kk], kb[kk],     s0);
#pragma unroll
    for (int kk = 0; kk < 4; ++kk) s1 = wmma_bf16(qfrag[kk], kb[4 + kk], s1);

    // V fragments issued early: latency overlaps the softmax VALU work.
    bf16x16 vb[8];
#pragma unroll
    for (int t = 0; t < 8; ++t)
      vb[t] = load_frag(vT, LPAD, h * HD + t * 16 + lane, base, half);

    int c0 = base + lane, c1 = c0 + 16;
    float rmax[8];
#pragma unroll
    for (int r = 0; r < 8; ++r) {
      float a = (c0 < L) ? s0[r] * scale : -3.0e38f;
      float b = (c1 < L) ? s1[r] * scale : -3.0e38f;
      s0[r] = a; s1[r] = b;
      rmax[r] = fmaxf(a, b);
    }
#pragma unroll
    for (int off = 1; off < 16; off <<= 1)
#pragma unroll
      for (int r = 0; r < 8; ++r)
        rmax[r] = fmaxf(rmax[r], __shfl_xor(rmax[r], off, 32));
    float corr[8], rsum[8];
#pragma unroll
    for (int r = 0; r < 8; ++r) {
      float mn = fmaxf(mrow[r], rmax[r]);
      corr[r] = __expf(mrow[r] - mn);
      mrow[r] = mn;
      float p0 = __expf(s0[r] - mn);
      float p1 = __expf(s1[r] - mn);
      lds[(r + 8 * half) * 32 + lane]      = p0;   // C-layout -> LDS [row][col]
      lds[(r + 8 * half) * 32 + 16 + lane] = p1;
      rsum[r] = p0 + p1;
    }
#pragma unroll
    for (int off = 1; off < 16; off <<= 1)
#pragma unroll
      for (int r = 0; r < 8; ++r)
        rsum[r] += __shfl_xor(rsum[r], off, 32);
#pragma unroll
    for (int r = 0; r < 8; ++r)
      lrow[r] = lrow[r] * corr[r] + rsum[r];
#pragma unroll
    for (int t = 0; t < 8; ++t)
#pragma unroll
      for (int r = 0; r < 8; ++r)
        o[t][r] *= corr[r];
    asm volatile("s_wait_dscnt 0x0" ::: "memory");  // stores visible before A-frag read
    FragU pf;                                        // P in A-fragment layout, bf16
#pragma unroll
    for (int i = 0; i < 8; ++i) {
      int k = ((i >> 2) << 4) + half * 8 + ((i & 3) << 1);
      pf.u[i] = pack2bf(lds[lane * 32 + k], lds[lane * 32 + k + 1]);
    }
#pragma unroll
    for (int t = 0; t < 8; ++t)                     // O += P * V  (8 d-tiles)
      o[t] = wmma_bf16(pf.b, vb[t], o[t]);
  }
  float inv[8];
#pragma unroll
  for (int r = 0; r < 8; ++r) inv[r] = 1.0f / lrow[r];
#pragma unroll
  for (int t = 0; t < 8; ++t) {
    int col = h * HD + t * 16 + lane;
#pragma unroll
    for (int r = 0; r < 8; ++r) {
      int row = qt * 16 + r + 8 * half;
      ob[(size_t)row * DIM + col] = f2bf(o[t][r] * inv[r]);
    }
  }
}

// ---------------- host-side orchestration ----------------
extern "C" void kernel_launch(void* const* d_in, const int* in_sizes, int n_in,
                              void* d_out, int out_size, void* d_ws, size_t ws_size,
                              hipStream_t stream) {
  const float* x     = (const float*)d_in[0];
  const float* freqs = (const float*)d_in[2];
  const float* ck    = (const float*)d_in[3];
  const float* cv    = (const float*)d_in[4];
  const float* Wq    = (const float*)d_in[5];
  const float* bq    = (const float*)d_in[6];
  const float* Wk    = (const float*)d_in[7];
  const float* bk    = (const float*)d_in[8];
  const float* Wv    = (const float*)d_in[9];
  const float* bv    = (const float*)d_in[10];
  const float* Wo    = (const float*)d_in[11];
  const float* bo    = (const float*)d_in[12];
  const float* gq    = (const float*)d_in[13];
  const float* gk    = (const float*)d_in[14];
  const int*   kvs   = (const int*)d_in[15];
  const int*   kvl   = (const int*)d_in[16];
  float* out = (float*)d_out;

  char* w = (char*)d_ws;
  size_t off = 0;
  auto alloc = [&](size_t bytes) -> char* {
    char* p = w + off;
    off = (off + bytes + 255) & ~(size_t)255;
    return p;
  };
  unsigned short* xb    = (unsigned short*)alloc((size_t)SEQ * DIM * 2);
  unsigned short* WqT   = (unsigned short*)alloc((size_t)DIM * DIM * 2);
  unsigned short* WkT   = (unsigned short*)alloc((size_t)DIM * DIM * 2);
  unsigned short* WvT   = (unsigned short*)alloc((size_t)DIM * DIM * 2);
  unsigned short* WoT   = (unsigned short*)alloc((size_t)DIM * DIM * 2);
  float*          Qf    = (float*)alloc((size_t)SEQ * DIM * 4);
  float*          Kf    = (float*)alloc((size_t)SEQ * DIM * 4);
  float*          Vf    = (float*)alloc((size_t)SEQ * DIM * 4);
  unsigned short* qb    = (unsigned short*)alloc((size_t)SEQ * DIM * 2);
  unsigned short* kfull = (unsigned short*)alloc((size_t)LPAD * DIM * 2);
  unsigned short* vT    = (unsigned short*)alloc((size_t)DIM * LPAD * 2);
  unsigned short* ob    = (unsigned short*)alloc((size_t)SEQ * DIM * 2);

  // 1) conversions / layout
  k_cvt_x<<<(SEQ * DIM + 255) / 256, 256, 0, stream>>>(x, xb, SEQ * DIM);
  k_transpose_w<<<(DIM * DIM + 255) / 256, 256, 0, stream>>>(Wq, WqT);
  k_transpose_w<<<(DIM * DIM + 255) / 256, 256, 0, stream>>>(Wk, WkT);
  k_transpose_w<<<(DIM * DIM + 255) / 256, 256, 0, stream>>>(Wv, WvT);
  k_transpose_w<<<(DIM * DIM + 255) / 256, 256, 0, stream>>>(Wo, WoT);
  k_cache<<<(LPAD * DIM + 255) / 256, 256, 0, stream>>>(ck, cv, kfull, vT, kvs, kvl);

  // 2) QKV projections (bf16 WMMA, f32 accumulate + bias)
  dim3 ggrid(DIM / (8 * 4 * 16), SEQ / 16);   // (4, 65)
  k_gemm<<<ggrid, 256, 0, stream>>>(xb, WqT, bq, Qf, DIM, DIM);
  k_gemm<<<ggrid, 256, 0, stream>>>(xb, WkT, bk, Kf, DIM, DIM);
  k_gemm<<<ggrid, 256, 0, stream>>>(xb, WvT, bv, Vf, DIM, DIM);

  // 3) RMSNorm + RoPE + append new KV
  k_norm_rope<<<SEQ, 256, 0, stream>>>(Qf, Kf, Vf, gq, gk, freqs, qb, kfull, vT, kvl);

  // 4) flash attention
  k_attn<<<dim3(NH, (QT + 3) / 4), 128, 0, stream>>>(qb, kfull, vT, ob, kvl);

  // 5) output projection -> f32 out
  k_gemm<<<ggrid, 256, 0, stream>>>(ob, WoT, bo, out, DIM, DIM);
  (void)in_sizes; (void)n_in; (void)out_size; (void)ws_size;
}
